// SpaMIE_joint_60885456388747
// MI455X (gfx1250) — matrix operand order repaired and checked by reading, not compiled, not run on hardware
//
#include <hip/hip_runtime.h>
#include <cstddef>

// ---------------------------------------------------------------------------
// SpaMIE joint forward for MI455X (gfx1250, wave32, WMMA).
// N=20000 nodes, E=320000 edges, IN=256, H=128, L=3.
// CSR gather aggregation (atomic-free) + fused normalize/f16-quantize +
// v_wmma_f32_16x16x32_f16 GEMMs with LDS-resident transposed f16 weights.
// K is a template parameter so A-fragments stay in VGPRs (no scratch spills).
// ---------------------------------------------------------------------------

#define NN 20000
#define NE 320000
#define D_IN 256
#define D_H  128

typedef __attribute__((ext_vector_type(16))) _Float16 v16h;
typedef __attribute__((ext_vector_type(8)))  _Float16 v8h;
typedef __attribute__((ext_vector_type(4)))  _Float16 v4h;
typedef __attribute__((ext_vector_type(8)))  float    v8f;
typedef __attribute__((ext_vector_type(4)))  float    f4;

// ---------------- CSR build ------------------------------------------------

__global__ __launch_bounds__(256) void k_zero_i32(int* p, int n) {
  int i = blockIdx.x * 256 + threadIdx.x;
  if (i < n) p[i] = 0;
}

__global__ __launch_bounds__(256) void k_hist(const int* __restrict__ dst, int e,
                                              int* __restrict__ deg) {
  int i = blockIdx.x * 256 + threadIdx.x;
  if (i < e) atomicAdd(&deg[dst[i]], 1);
}

// Single-workgroup chunked exclusive scan (N=20000, cheap, runs once/graph).
__global__ __launch_bounds__(1024) void k_scan(const int* __restrict__ deg,
                                               int* __restrict__ row_start,
                                               int* __restrict__ cursor, int n) {
  __shared__ int buf[1024];
  __shared__ int carry;
  if (threadIdx.x == 0) carry = 0;
  __syncthreads();
  for (int base = 0; base < n; base += 1024) {
    int i = base + (int)threadIdx.x;
    int v = (i < n) ? deg[i] : 0;
    buf[threadIdx.x] = v;
    __syncthreads();
    for (int off = 1; off < 1024; off <<= 1) {
      int t = (threadIdx.x >= (unsigned)off) ? buf[threadIdx.x - off] : 0;
      __syncthreads();
      buf[threadIdx.x] += t;
      __syncthreads();
    }
    int excl = buf[threadIdx.x] - v;
    if (i < n) { row_start[i] = carry + excl; cursor[i] = carry + excl; }
    __syncthreads();
    if (threadIdx.x == 0) carry += buf[1023];
    __syncthreads();
  }
  if (threadIdx.x == 0) row_start[n] = carry;
}

__global__ __launch_bounds__(256) void k_fill(const int* __restrict__ src,
                                              const int* __restrict__ dst, int e,
                                              int* __restrict__ cursor,
                                              int* __restrict__ csr_src) {
  int i = blockIdx.x * 256 + threadIdx.x;
  if (i < e) {
    int p = atomicAdd(&cursor[dst[i]], 1);
    csr_src[p] = src[i];
  }
}

// ---------------- SAGE aggregation (gather, normalize, quantize) -----------
// h16[i,:] = f16( (x[i,:] + sum_{j in N(i)} x[j,:]) / (deg_i + 1) )
// D/4 lanes per node, float4 per lane; neighbor rows live in L2 (table<=20MB).
template <int D>
__global__ __launch_bounds__(256) void k_aggregate(const float* __restrict__ x,
                                                   const int* __restrict__ row_start,
                                                   const int* __restrict__ csr_src,
                                                   _Float16* __restrict__ h16,
                                                   int n) {
  constexpr int TPN = D >> 2;                          // 32 or 64 lanes/node
  int lane = (int)threadIdx.x & (TPN - 1);
  int nidx = (int)(blockIdx.x * 256 + threadIdx.x) >> (D == 256 ? 6 : 5);
  if (nidx >= n) return;
  int c0 = lane << 2;
  f4 acc = *(const f4*)(x + (size_t)nidx * D + c0);
  int s0 = row_start[nidx], s1 = row_start[nidx + 1];
  for (int e = s0; e < s1; ++e) {
    int s = csr_src[e];
    if (e + 1 < s1)
      __builtin_prefetch(x + (size_t)csr_src[e + 1] * D + c0, 0, 0);
    acc += *(const f4*)(x + (size_t)s * D + c0);
  }
  float inv = 1.0f / (float)(s1 - s0 + 1);
  acc = acc * inv;
  v4h hv;
  hv.x = (_Float16)acc.x; hv.y = (_Float16)acc.y;
  hv.z = (_Float16)acc.z; hv.w = (_Float16)acc.w;
  *(v4h*)(h16 + (size_t)nidx * D + c0) = hv;
}

// ---------------- weight prep ----------------------------------------------
// WT[m*K + k] = (f16) W[k*M + m]   (transpose so B-columns are contiguous)
__global__ __launch_bounds__(256) void k_cvt_wt(const float* __restrict__ W,
                                                _Float16* __restrict__ WT,
                                                int K, int M) {
  int i = blockIdx.x * 256 + threadIdx.x;
  if (i < K * M) {
    int m = i / K, k = i - m * K;
    WT[i] = (_Float16)W[(size_t)k * M + m];
  }
}

__global__ __launch_bounds__(256) void k_cvt_h(const float* __restrict__ s,
                                               _Float16* __restrict__ d, int n) {
  int i = blockIdx.x * 256 + threadIdx.x;
  if (i < n) d[i] = (_Float16)s[i];
}

// ---------------- WMMA GEMM ------------------------------------------------
// C[n,M] = A[n,K](f16) * WT^T + bias, optional tanh epilogue.
// Block: 256 threads = 8 waves; block tile 128 rows x 128 cols (grid.y slabs).
// WT slab staged in LDS with +8-half row pad (kills ds bank conflicts).
// K is compile-time so all fragment arrays live in VGPRs (no dynamic index).
template <int K>
__global__ __launch_bounds__(256) void k_gemm_wmma(const _Float16* __restrict__ A,
                                                   const _Float16* __restrict__ WT,
                                                   const float* __restrict__ bias,
                                                   float* __restrict__ C,
                                                   int n, int M, int act) {
  extern __shared__ _Float16 lds[];
  constexpr int KF   = K >> 5;                      // 16x16x32 k-steps (4 or 8)
  constexpr int LDSK = K + 8;
  const int tid = (int)threadIdx.x;
  const int n0  = (int)blockIdx.y * 128;
  {
    constexpr int CPR = K >> 3;                     // v8h chunks per row
#pragma unroll 4
    for (int i = tid; i < 128 * CPR; i += 256) {
      int r = i / CPR, c = i - r * CPR;
      *(v8h*)(lds + (size_t)r * LDSK + (size_t)c * 8) =
          *(const v8h*)(WT + (size_t)(n0 + r) * K + (size_t)c * 8);
    }
  }
  __syncthreads();

  const int wave  = tid >> 5;
  const int lane  = tid & 31;
  const int lhalf = lane >> 4;                      // 0: lanes 0-15, 1: 16-31
  const int l16   = lane & 15;
  const int col   = n0 + wave * 16 + l16;
  const float bv  = bias ? bias[col] : 0.0f;

  for (int mt = 0; mt < 8; ++mt) {
    int m_base = (int)blockIdx.x * 128 + mt * 16;
    if (m_base >= n) break;                         // uniform: N % 16 == 0
    // A fragments: elem 0..7 = A[row, k*32 + lhalf*8 + 0..7], elem 8..15 = +16
    const _Float16* ap = A + (size_t)(m_base + l16) * K + lhalf * 8;
    union { v16h v; v8h h[2]; } af[KF];
#pragma unroll
    for (int k = 0; k < KF; ++k) {
      af[k].h[0] = *(const v8h*)(ap + k * 32);
      af[k].h[1] = *(const v8h*)(ap + k * 32 + 16);
    }
    v8f c;
#pragma unroll
    for (int r = 0; r < 8; ++r) c[r] = bv;
    // B fragments: elem 0..7 = K kb + lhalf*8 + 0..7, elem 8..15 = +16 (N=l16)
    const _Float16* bp = lds + (size_t)(wave * 16 + l16) * LDSK + lhalf * 8;
#pragma unroll
    for (int k = 0; k < KF; ++k) {
      union { v16h v; v8h h[2]; } bf;
      bf.h[0] = *(const v8h*)(bp + k * 32);
      bf.h[1] = *(const v8h*)(bp + k * 32 + 16);
      c = __builtin_amdgcn_wmma_f32_16x16x32_f16(false, af[k].v, false, bf.v,
                                                 (short)0, c, false, false);
    }
    float* cp = C + (size_t)(m_base + lhalf * 8) * M + col;
#pragma unroll
    for (int r = 0; r < 8; ++r) {
      float vv = c[r];
      if (act) vv = tanhf(vv);
      cp[(size_t)r * M] = vv;
    }
  }
}

// ---------------- combine + attention --------------------------------------

__global__ __launch_bounds__(256) void k_combine(const float* __restrict__ hs,
                                                 const float* __restrict__ wt,
                                                 float* __restrict__ out,
                                                 float* __restrict__ wt_out,
                                                 int nh) {
  float w0 = wt[0], w1 = wt[1], w2 = wt[2];
  float mx = fmaxf(w0, fmaxf(w1, w2));
  float e0 = __expf(w0 - mx), e1 = __expf(w1 - mx), e2 = __expf(w2 - mx);
  float inv = 1.0f / (e0 + e1 + e2);
  e0 *= inv; e1 *= inv; e2 *= inv;
  int i = blockIdx.x * 256 + threadIdx.x;
  if (i < nh)
    out[i] = e0 * hs[i] + e1 * hs[(size_t)nh + i] + e2 * hs[2 * (size_t)nh + i];
  if (blockIdx.x == 0 && threadIdx.x == 0) {
    wt_out[0] = e0; wt_out[1] = e1; wt_out[2] = e2;
  }
}

// one wave32 per node: vu = v@u, alpha = softmax, out = a0*e1 + a1*e2
__global__ __launch_bounds__(256) void k_attn_blend(const float* __restrict__ e1,
                                                    const float* __restrict__ e2,
                                                    const float* __restrict__ v1,
                                                    const float* __restrict__ v2,
                                                    const float* __restrict__ u,
                                                    float* __restrict__ out,
                                                    float* __restrict__ alph,
                                                    int n, int h) {
  int gid  = (int)(blockIdx.x * 256 + threadIdx.x);
  int node = gid >> 5;
  int lane = gid & 31;
  if (node >= n) return;
  float p1 = 0.0f, p2 = 0.0f;
  for (int c = lane; c < h; c += 32) {
    float uc = u[c];
    p1 += v1[(size_t)node * h + c] * uc;
    p2 += v2[(size_t)node * h + c] * uc;
  }
  for (int off = 16; off > 0; off >>= 1) {
    p1 += __shfl_xor(p1, off);
    p2 += __shfl_xor(p2, off);
  }
  float a1 = p1 + 1e-6f, a2 = p2 + 1e-6f;
  float mx = fmaxf(a1, a2);
  float x1 = __expf(a1 - mx), x2 = __expf(a2 - mx);
  float inv = 1.0f / (x1 + x2);
  float w1 = x1 * inv, w2 = x2 * inv;
  if (alph && lane == 0) { alph[(size_t)node * 2] = w1; alph[(size_t)node * 2 + 1] = w2; }
  for (int c = lane; c < h; c += 32)
    out[(size_t)node * h + c] =
        w1 * e1[(size_t)node * h + c] + w2 * e2[(size_t)node * h + c];
}

// ---------------------------------------------------------------------------

extern "C" void kernel_launch(void* const* d_in, const int* in_sizes, int n_in,
                              void* d_out, int out_size, void* d_ws, size_t ws_size,
                              hipStream_t stream) {
  (void)in_sizes; (void)n_in; (void)out_size; (void)ws_size;
  const int N = NN, E = NE;

  // ---- inputs -------------------------------------------------------------
  const int* srcp[4] = {(const int*)d_in[0], (const int*)d_in[2],
                        (const int*)d_in[4], (const int*)d_in[6]};
  const int* dstp[4] = {(const int*)d_in[1], (const int*)d_in[3],
                        (const int*)d_in[5], (const int*)d_in[7]};
  const float* feat1 = (const float*)d_in[8];
  const float* feat2 = (const float*)d_in[9];
  const float* enc1_W0 = (const float*)d_in[10]; const float* enc1_b0 = (const float*)d_in[11];
  const float* enc1_W  = (const float*)d_in[12]; const float* enc1_b  = (const float*)d_in[13];
  const float* enc2_W0 = (const float*)d_in[14]; const float* enc2_b0 = (const float*)d_in[15];
  const float* enc2_W  = (const float*)d_in[16]; const float* enc2_b  = (const float*)d_in[17];
  const float* dec1_W0 = (const float*)d_in[18]; const float* dec1_b0 = (const float*)d_in[19];
  const float* dec1_W  = (const float*)d_in[20]; const float* dec1_b  = (const float*)d_in[21];
  const float* dec2_W0 = (const float*)d_in[22]; const float* dec2_b0 = (const float*)d_in[23];
  const float* dec2_W  = (const float*)d_in[24]; const float* dec2_b  = (const float*)d_in[25];
  const float* att1_w = (const float*)d_in[26]; const float* att1_u = (const float*)d_in[27];
  const float* att2_w = (const float*)d_in[28]; const float* att2_u = (const float*)d_in[29];
  const float* attc_w = (const float*)d_in[30]; const float* attc_u = (const float*)d_in[31];
  const float* wt1 = (const float*)d_in[32]; const float* wt2 = (const float*)d_in[33];
  const float* wt3 = (const float*)d_in[34]; const float* wt4 = (const float*)d_in[35];

  // ---- workspace carve (~171 MB) -----------------------------------------
  char* p = (char*)d_ws;
  auto alloc = [&](size_t bytes) -> char* {
    char* r = p; p += (bytes + 255) & ~(size_t)255; return r;
  };
  int *deg[4], *rs[4], *cur[4], *csr[4];
  for (int g = 0; g < 4; ++g) {
    deg[g] = (int*)alloc((size_t)N * 4);
    rs[g]  = (int*)alloc((size_t)(N + 1) * 4);
    cur[g] = (int*)alloc((size_t)N * 4);
    csr[g] = (int*)alloc((size_t)E * 4);
  }
  _Float16 *wt_e1[3], *wt_e2[3], *wt_d1[3], *wt_d2[3];
  wt_e1[0] = (_Float16*)alloc(D_H * D_IN * 2);
  wt_e1[1] = (_Float16*)alloc(D_H * D_H * 2);
  wt_e1[2] = (_Float16*)alloc(D_H * D_H * 2);
  wt_e2[0] = (_Float16*)alloc(D_H * D_IN * 2);
  wt_e2[1] = (_Float16*)alloc(D_H * D_H * 2);
  wt_e2[2] = (_Float16*)alloc(D_H * D_H * 2);
  wt_d1[0] = (_Float16*)alloc(D_IN * D_H * 2);
  wt_d1[1] = (_Float16*)alloc(D_IN * D_IN * 2);
  wt_d1[2] = (_Float16*)alloc(D_IN * D_IN * 2);
  wt_d2[0] = (_Float16*)alloc(D_IN * D_H * 2);
  wt_d2[1] = (_Float16*)alloc(D_IN * D_IN * 2);
  wt_d2[2] = (_Float16*)alloc(D_IN * D_IN * 2);
  _Float16* w16_a1 = (_Float16*)alloc(D_H * D_H * 2);
  _Float16* w16_a2 = (_Float16*)alloc(D_H * D_H * 2);
  _Float16* w16_ac = (_Float16*)alloc(D_H * D_H * 2);
  _Float16* h16  = (_Float16*)alloc((size_t)N * D_IN * 2);
  _Float16* e16a = (_Float16*)alloc((size_t)N * D_H * 2);
  _Float16* e16b = (_Float16*)alloc((size_t)N * D_H * 2);
  const size_t NHs = (size_t)N * D_H;
  const size_t NIs = (size_t)N * D_IN;
  float* stackA = (float*)alloc(3 * NHs * 4);
  float* stackB = (float*)alloc(3 * NHs * 4);
  float* eA = (float*)alloc(NHs * 4);
  float* eB = (float*)alloc(NHs * 4);
  float* v1 = (float*)alloc(NHs * 4);
  float* v2 = (float*)alloc(NHs * 4);
  float* t0 = (float*)alloc(NIs * 4);
  float* t1 = (float*)alloc(NIs * 4);

  // ---- output layout (flat tuple order) -----------------------------------
  float* O = (float*)d_out;
  float* o_embc = O;                 // [N,H]
  float* o_wt   = o_embc + NHs;      // [4,3]
  float* o_d1   = o_wt + 12;         // [N,IN]
  float* o_d2   = o_d1 + NIs;        // [N,IN]
  float* o_emb1 = o_d2 + NIs;        // [N,H]
  float* o_emb2 = o_emb1 + NHs;      // [N,H]
  float* o_cr1  = o_emb2 + NHs;      // [3,N,H]
  float* o_cr2  = o_cr1 + 3 * NHs;   // [3,N,H]
  float* o_alph = o_cr2 + 3 * NHs;   // [N,2]

  // ---- CSR build (4 graphs) ----------------------------------------------
  for (int g = 0; g < 4; ++g) {
    k_zero_i32<<<(N + 255) / 256, 256, 0, stream>>>(deg[g], N);
    k_hist<<<(E + 255) / 256, 256, 0, stream>>>(dstp[g], E, deg[g]);
    k_scan<<<1, 1024, 0, stream>>>(deg[g], rs[g], cur[g], N);
    k_fill<<<(E + 255) / 256, 256, 0, stream>>>(srcp[g], dstp[g], E, cur[g], csr[g]);
  }

  // ---- weight transpose+f16 conversion (once per launch) ------------------
  auto cvt = [&](const float* W, _Float16* WT, int K, int M) {
    int tot = K * M;
    k_cvt_wt<<<(tot + 255) / 256, 256, 0, stream>>>(W, WT, K, M);
  };
  cvt(enc1_W0, wt_e1[0], D_IN, D_H);
  cvt(enc1_W,               wt_e1[1], D_H, D_H);
  cvt(enc1_W + D_H * D_H,   wt_e1[2], D_H, D_H);
  cvt(enc2_W0, wt_e2[0], D_IN, D_H);
  cvt(enc2_W,               wt_e2[1], D_H, D_H);
  cvt(enc2_W + D_H * D_H,   wt_e2[2], D_H, D_H);
  cvt(dec1_W0, wt_d1[0], D_H, D_IN);
  cvt(dec1_W,               wt_d1[1], D_IN, D_IN);
  cvt(dec1_W + D_IN * D_IN, wt_d1[2], D_IN, D_IN);
  cvt(dec2_W0, wt_d2[0], D_H, D_IN);
  cvt(dec2_W,               wt_d2[1], D_IN, D_IN);
  cvt(dec2_W + D_IN * D_IN, wt_d2[2], D_IN, D_IN);
  cvt(att1_w, w16_a1, D_H, D_H);
  cvt(att2_w, w16_a2, D_H, D_H);
  cvt(attc_w, w16_ac, D_H, D_H);

  // ---- building blocks ----------------------------------------------------
  auto sage = [&](int g, const float* xin, int Din, const _Float16* WT,
                  const float* b, float* out, int Dout) {
    dim3 gg((N + 127) / 128, Dout / 128);
    if (Din == 256) {
      k_aggregate<256><<<5000, 256, 0, stream>>>(xin, rs[g], csr[g], h16, N);
      size_t lds = (size_t)128 * (256 + 8) * sizeof(_Float16);
      k_gemm_wmma<256><<<gg, 256, lds, stream>>>(h16, WT, b, out, N, Dout, 0);
    } else {
      k_aggregate<128><<<2500, 256, 0, stream>>>(xin, rs[g], csr[g], h16, N);
      size_t lds = (size_t)128 * (128 + 8) * sizeof(_Float16);
      k_gemm_wmma<128><<<gg, 256, lds, stream>>>(h16, WT, b, out, N, Dout, 0);
    }
  };
  auto run_enc = [&](int g, const float* xin, _Float16* const* WT,
                     const float* b0, const float* bs, float* stack) {
    sage(g, xin, D_IN, WT[0], b0, stack, D_H);
    sage(g, stack, D_H, WT[1], bs, stack + NHs, D_H);
    sage(g, stack + NHs, D_H, WT[2], bs + D_H, stack + 2 * NHs, D_H);
  };
  auto run_dec = [&](int g, const float* xin, _Float16* const* WT,
                     const float* b0, const float* bs, float* out) {
    sage(g, xin, D_H, WT[0], b0, t0, D_IN);
    sage(g, t0, D_IN, WT[1], bs, t1, D_IN);
    sage(g, t1, D_IN, WT[2], bs + D_IN, out, D_IN);
  };
  auto comb = [&](const float* stack, const float* wtv, float* out, float* wrow) {
    int nh = N * D_H;
    k_combine<<<(nh + 255) / 256, 256, 0, stream>>>(stack, wtv, out, wrow, nh);
  };
  auto attn = [&](const float* x1, const float* x2, const _Float16* w16,
                  const float* u, float* out, float* alph) {
    int nh = N * D_H;
    k_cvt_h<<<(nh + 255) / 256, 256, 0, stream>>>(x1, e16a, nh);
    k_cvt_h<<<(nh + 255) / 256, 256, 0, stream>>>(x2, e16b, nh);
    dim3 gg((N + 127) / 128, 1);
    size_t lds = (size_t)128 * (128 + 8) * sizeof(_Float16);
    k_gemm_wmma<128><<<gg, 256, lds, stream>>>(e16a, w16, nullptr, v1, N, D_H, 1);
    k_gemm_wmma<128><<<gg, 256, lds, stream>>>(e16b, w16, nullptr, v2, N, D_H, 1);
    k_attn_blend<<<(N * 32 + 255) / 256, 256, 0, stream>>>(x1, x2, v1, v2, u,
                                                           out, alph, N, D_H);
  };

  // ---- forward graph ------------------------------------------------------
  run_enc(0, feat1, wt_e1, enc1_b0, enc1_b, stackA);        // x_sp1
  run_enc(1, feat1, wt_e1, enc1_b0, enc1_b, stackB);        // x_fe1
  comb(stackA, wt1, eA, o_wt + 0);
  comb(stackB, wt2, eB, o_wt + 3);
  attn(eA, eB, w16_a1, att1_u, o_emb1, nullptr);            // emb1

  run_enc(2, feat2, wt_e2, enc2_b0, enc2_b, stackA);        // x_sp2
  run_enc(3, feat2, wt_e2, enc2_b0, enc2_b, stackB);        // x_fe2
  comb(stackA, wt3, eA, o_wt + 6);
  comb(stackB, wt4, eB, o_wt + 9);
  attn(eA, eB, w16_a2, att2_u, o_emb2, nullptr);            // emb2

  attn(o_emb1, o_emb2, w16_ac, attc_u, o_embc, o_alph);     // emb_comb, alph

  run_dec(0, o_embc, wt_d1, dec1_b0, dec1_b, o_d1);         // d1
  run_dec(0, o_embc, wt_d2, dec2_b0, dec2_b, o_d2);         // d2

  float* tmpdec = stackA;                                   // [N,IN] scratch
  run_dec(0, o_emb1, wt_d2, dec2_b0, dec2_b, tmpdec);       // dec2(e1s, emb1)
  run_enc(0, tmpdec, wt_e2, enc2_b0, enc2_b, o_cr1);        // cross1 stack
  run_dec(2, o_emb2, wt_d1, dec1_b0, dec1_b, tmpdec);       // dec1(e2s, emb2)
  run_enc(2, tmpdec, wt_e1, enc1_b0, enc1_b, o_cr2);        // cross2 stack
}
